// MultiHeadAttention_47863115546638
// MI455X (gfx1250) — matrix operand order repaired and checked by reading, not compiled, not run on hardware
//
#include <hip/hip_runtime.h>
#include <hip/hip_bf16.h>

// Problem constants (from reference)
#define B_  8
#define S_  1024
#define D_  1024
#define H_  16
#define DKV 64
static constexpr float NEGV = -1e10f;

typedef __attribute__((ext_vector_type(16))) __bf16        v16bf;
typedef __attribute__((ext_vector_type(8)))  float         v8f;
typedef __attribute__((ext_vector_type(8)))  unsigned int  v8u;
typedef __attribute__((ext_vector_type(4)))  unsigned int  u32x4;
typedef __attribute__((ext_vector_type(8)))  int           i32x8;
typedef __attribute__((ext_vector_type(4)))  int           i32x4;

__device__ __forceinline__ unsigned short f2bf(float f) {
  unsigned u = __builtin_bit_cast(unsigned, f);
  u += 0x7FFFu + ((u >> 16) & 1u);           // round-to-nearest-even
  return (unsigned short)(u >> 16);
}

__device__ __forceinline__ v8f v8f_zero() {
  v8f z;
  #pragma unroll
  for (int e = 0; e < 8; ++e) z[e] = 0.f;
  return z;
}

// LDS byte-offset of a __shared__ object (generic -> AS3 -> int)
typedef __attribute__((address_space(3))) const unsigned char lds_cbyte;
__device__ __forceinline__ unsigned lds_offset(const void* p) {
  return (unsigned)(unsigned long long)(lds_cbyte*)p;
}

// ---------------------------------------------------------------------------
// TDM 2D tile load: global (row-major, stride_elems bf16/row) -> LDS, with
// per-row padding so LDS row stride = rowlen + pad.  Descriptor per ISA §8.
//   pad_interval_code: row payload DWORDs = 2 << code
//   pad_amount_code  : pad DWORDs = code + 1
// ---------------------------------------------------------------------------
__device__ __forceinline__ void tdm_load_2d(unsigned lds_off, const void* gptr,
                                            unsigned rows, unsigned rowlen,
                                            unsigned stride_elems,
                                            unsigned pad_interval_code,
                                            unsigned pad_amount_code)
{
  unsigned long long ga = (unsigned long long)gptr;
  u32x4 g0;
  g0[0] = 1u;                                            // count=1 (valid), user mode
  g0[1] = lds_off;                                       // lds_addr
  g0[2] = (unsigned)(ga & 0xFFFFFFFFu);                  // global_addr[31:0]
  g0[3] = (unsigned)((ga >> 32) & 0x01FFFFFFu)           // global_addr[56:32]
        | (2u << 30);                                    // type=2 ("image")
  i32x8 g1;
  g1[0] = (int)((1u << 16)                               // data_size = 2 bytes
              | (1u << 20)                               // pad_enable
              | (pad_interval_code << 22)
              | (pad_amount_code << 25));
  g1[1] = (int)((rowlen & 0xFFFFu) << 16);               // tensor_dim0[15:0]
  g1[2] = (int)(((rowlen >> 16) & 0xFFFFu)               // tensor_dim0[31:16]
              | ((rows & 0xFFFFu) << 16));               // tensor_dim1[15:0]
  g1[3] = (int)(((rows >> 16) & 0xFFFFu)                 // tensor_dim1[31:16]
              | ((rowlen & 0xFFFFu) << 16));             // tile_dim0
  g1[4] = (int)(rows & 0xFFFFu);                         // tile_dim1 (tile_dim2=0)
  g1[5] = (int)stride_elems;                             // tensor_dim0_stride[31:0]
  g1[6] = 0;                                             // stride hi / dim1_stride lo
  g1[7] = 0;
  i32x4 z4; z4[0]=0; z4[1]=0; z4[2]=0; z4[3]=0;
  i32x8 z8;
  #pragma unroll
  for (int e = 0; e < 8; ++e) z8[e] = 0;
  __builtin_amdgcn_tensor_load_to_lds(g0, g1, z4, z4, z8, 0);
}

// ---------------------------------------------------------------------------
// Prep 1: elementwise f32 -> bf16 (vectorized float4 -> 4x bf16)
// ---------------------------------------------------------------------------
__global__ __launch_bounds__(256)
void f32_to_bf16_kernel(const float* __restrict__ src,
                        unsigned short* __restrict__ dst, int n4)
{
  int i = blockIdx.x * blockDim.x + threadIdx.x;
  if (i < n4) {
    float4 v = ((const float4*)src)[i];
    unsigned lo = ((unsigned)f2bf(v.y) << 16) | f2bf(v.x);
    unsigned hi = ((unsigned)f2bf(v.w) << 16) | f2bf(v.z);
    ((uint2*)dst)[i] = make_uint2(lo, hi);
  }
}

// ---------------------------------------------------------------------------
// Prep 2: weight transpose+convert.  W [1024 out][1024 in] f32 ->
// Wt [1024 in(K)][1024 out(N)] bf16.  64x64 LDS tile per block, z selects W.
// ---------------------------------------------------------------------------
__global__ __launch_bounds__(256)
void wt_transpose_kernel(const float* __restrict__ W0, const float* __restrict__ W1,
                         const float* __restrict__ W2, const float* __restrict__ W3,
                         unsigned short* __restrict__ T0, unsigned short* __restrict__ T1,
                         unsigned short* __restrict__ T2, unsigned short* __restrict__ T3)
{
  const float* W = (blockIdx.z == 0) ? W0 : (blockIdx.z == 1) ? W1
                 : (blockIdx.z == 2) ? W2 : W3;
  unsigned short* T = (blockIdx.z == 0) ? T0 : (blockIdx.z == 1) ? T1
                    : (blockIdx.z == 2) ? T2 : T3;
  __shared__ __align__(16) unsigned short Ts[64 * 72];
  const int t  = threadIdx.x;
  const int k0 = blockIdx.x * 64;
  const int n0 = blockIdx.y * 64;
  #pragma unroll
  for (int p = 0; p < 4; ++p) {
    int r  = (t >> 4) + 16 * p;                // out-row within tile
    int c4 = (t & 15) * 4;                     // in-col quad
    float4 v = *(const float4*)&W[(size_t)(n0 + r) * 1024 + k0 + c4];
    unsigned lo = ((unsigned)f2bf(v.y) << 16) | f2bf(v.x);
    unsigned hi = ((unsigned)f2bf(v.w) << 16) | f2bf(v.z);
    *(uint2*)&Ts[r * 72 + c4] = make_uint2(lo, hi);
  }
  __syncthreads();
  // output row = k, gather Ts[n_local][k]
  int c   = t >> 2;                            // k within tile (0..63)
  int seg = (t & 3) * 16;                      // n segment
  unsigned pk[4];
  #pragma unroll
  for (int g = 0; g < 4; ++g) {
    unsigned short e0 = Ts[(seg + 2*g + 0) * 72 + c];
    unsigned short e1 = Ts[(seg + 2*g + 1) * 72 + c];
    pk[g] = ((unsigned)e1 << 16) | e0;
  }
  uint4 outA = make_uint4(pk[0], pk[1], pk[2], pk[3]);
  #pragma unroll
  for (int g = 0; g < 4; ++g) {
    unsigned short e0 = Ts[(seg + 8 + 2*g + 0) * 72 + c];
    unsigned short e1 = Ts[(seg + 8 + 2*g + 1) * 72 + c];
    pk[g] = ((unsigned)e1 << 16) | e0;
  }
  uint4 outB = make_uint4(pk[0], pk[1], pk[2], pk[3]);
  unsigned short* dst = &T[(size_t)(k0 + c) * 1024 + n0 + seg];
  *(uint4*)dst       = outA;
  *(uint4*)(dst + 8) = outB;
}

// ---------------------------------------------------------------------------
// Fragment helpers (ISA 16-bit A / B layouts)
// ---------------------------------------------------------------------------
__device__ __forceinline__ v16bf load_a_frag(const unsigned short* __restrict__ base,
                                             int lda, int m, int ks, int half)
{
  v8u au;
  #pragma unroll
  for (int i = 0; i < 8; ++i) {
    int kb = ks + (i >> 2) * 16 + half * 8 + (i & 3) * 2;
    au[i] = *(const unsigned*)&base[m * lda + kb];
  }
  return __builtin_bit_cast(v16bf, au);
}

__device__ __forceinline__ v16bf load_b_frag(const unsigned short* __restrict__ src,
                                             int ldb, int ks, int half)
{
  v8u bu;
  #pragma unroll
  for (int i = 0; i < 8; ++i) {
    int kb = ks + (i >> 2) * 16 + half * 8 + (i & 3) * 2;
    bu[i] = (unsigned)src[kb * ldb] | ((unsigned)src[(kb + 1) * ldb] << 16);
  }
  return __builtin_bit_cast(v16bf, bu);
}

// ---------------------------------------------------------------------------
// GEMM: C[M,N] = A[M,K] * Bk[K,N], both bf16 row-major in global.
// TDM double-buffered tile staging.  Tile 128x128x64, 8 waves, wave 32x64.
// OUT_MODE: 0 = bf16 row-major, 1 = bf16 per-head transposed (khT), 2 = f32.
// ---------------------------------------------------------------------------
template <int OUT_MODE>
__global__ __launch_bounds__(256)
void gemm_tdm_wmma(const unsigned short* __restrict__ A,
                   const unsigned short* __restrict__ Bk,
                   void* __restrict__ Cp, int M, int N, int K)
{
  constexpr int BM = 128, BN = 128, BK = 64;
  constexpr int LDA = BK + 8;   // 72  : row 128B + 16B pad
  constexpr int LDB = BN + 8;   // 136 : row 256B + 16B pad
  __shared__ __align__(16) unsigned short As[2][BM * LDA];
  __shared__ __align__(16) unsigned short Bs[2][BK * LDB];

  const int t    = threadIdx.x;
  const int w    = t >> 5;
  const int lane = t & 31;
  const int half = lane >> 4;
  const int l16  = lane & 15;
  const int wm   = w >> 1;
  const int wn   = w & 1;
  const int m0   = blockIdx.y * BM;
  const int n0   = blockIdx.x * BN;
  const int T    = K / BK;

  v8f acc[2][4];
  #pragma unroll
  for (int i = 0; i < 2; ++i)
    #pragma unroll
    for (int j = 0; j < 4; ++j) acc[i][j] = v8f_zero();

  if (t < 32) {  // wave 0 drives the TDM
    tdm_load_2d(lds_offset(&As[0][0]), &A[(size_t)m0 * K],            BM, BK, K, 4, 3);
    tdm_load_2d(lds_offset(&Bs[0][0]), &Bk[(size_t)0 * N + n0],       BK, BN, N, 5, 3);
  }

  for (int tile = 0; tile < T; ++tile) {
    const int buf = tile & 1;
    if (t < 32) {
      if (tile + 1 < T) {
        int kt = (tile + 1) * BK;
        tdm_load_2d(lds_offset(&As[buf ^ 1][0]), &A[(size_t)m0 * K + kt],      BM, BK, K, 4, 3);
        tdm_load_2d(lds_offset(&Bs[buf ^ 1][0]), &Bk[(size_t)kt * N + n0],     BK, BN, N, 5, 3);
        __builtin_amdgcn_s_wait_tensorcnt(2);   // current tile's pair done
      } else {
        __builtin_amdgcn_s_wait_tensorcnt(0);
      }
    }
    __syncthreads();                            // tile `tile` visible to all

    const unsigned short* Asb = &As[buf][0];
    const unsigned short* Bsb = &Bs[buf][0];
    #pragma unroll
    for (int ks = 0; ks < BK; ks += 32) {
      v16bf afrag[2];
      #pragma unroll
      for (int fm = 0; fm < 2; ++fm)
        afrag[fm] = load_a_frag(Asb, LDA, wm * 32 + fm * 16 + l16, ks, half);
      v16bf bfrag[4];
      #pragma unroll
      for (int fn = 0; fn < 4; ++fn)
        bfrag[fn] = load_b_frag(&Bsb[wn * 64 + fn * 16 + l16], LDB, ks, half);
      #pragma unroll
      for (int fm = 0; fm < 2; ++fm)
        #pragma unroll
        for (int fn = 0; fn < 4; ++fn)
          acc[fm][fn] = __builtin_amdgcn_wmma_f32_16x16x32_bf16(
              false, afrag[fm], false, bfrag[fn], (short)0, acc[fm][fn],
              false, false);
    }
    __syncthreads();                            // consumers done before overwrite
  }

  // ---- epilogue ----
  #pragma unroll
  for (int fm = 0; fm < 2; ++fm) {
    #pragma unroll
    for (int fn = 0; fn < 4; ++fn) {
      int n = n0 + wn * 64 + fn * 16 + l16;
      if constexpr (OUT_MODE == 1) {
        // khT[b][h][dk][s], 8 consecutive s per lane -> packed uint4
        int hh = n >> 6, dk = n & 63;
        int mbase = m0 + wm * 32 + fm * 16 + 8 * half;
        int bb = mbase >> 10, ss = mbase & 1023;
        unsigned pk[4];
        #pragma unroll
        for (int g = 0; g < 4; ++g) {
          unsigned short e0 = f2bf(acc[fm][fn][2 * g + 0]);
          unsigned short e1 = f2bf(acc[fm][fn][2 * g + 1]);
          pk[g] = ((unsigned)e1 << 16) | e0;
        }
        unsigned short* dst = (unsigned short*)Cp
            + (((size_t)bb * H_ + hh) * DKV + dk) * S_ + ss;
        *(uint4*)dst = make_uint4(pk[0], pk[1], pk[2], pk[3]);
      } else {
        #pragma unroll
        for (int v = 0; v < 8; ++v) {
          int m = m0 + wm * 32 + fm * 16 + v + 8 * half;
          float val = acc[fm][fn][v];
          if constexpr (OUT_MODE == 0)
            ((unsigned short*)Cp)[(size_t)m * N + n] = f2bf(val);
          else
            ((float*)Cp)[(size_t)m * N + n] = val;
        }
      }
    }
  }
}

// ---------------------------------------------------------------------------
// Flash attention with TDM double-buffered K/V tiles.
// Block = (b*H + h, 128-row query tile); wave owns 16 query rows.
// Loops ALL key tiles so fully-masked rows match reference softmax exactly.
// ---------------------------------------------------------------------------
__global__ __launch_bounds__(256)
void flash_attn_wmma(const unsigned short* __restrict__ qh,
                     const unsigned short* __restrict__ khT,  // [B*H][DK][S]
                     const unsigned short* __restrict__ vh,   // [B*S][H*DV]
                     const int* __restrict__ pm1p, const int* __restrict__ pm2p,
                     unsigned short* __restrict__ ob)
{
  constexpr int LDQ = 72, LDT = 72, LDV = 72, LDP = 72;
  __shared__ __align__(16) unsigned short Qs[128 * LDQ];
  __shared__ __align__(16) unsigned short Kts[2][64 * LDT];   // [dk][j]
  __shared__ __align__(16) unsigned short Vs[2][64 * LDV];    // [j][dv]
  __shared__ __align__(16) unsigned short Ps[8 * 16 * LDP];

  const int t    = threadIdx.x;
  const int w    = t >> 5;
  const int lane = t & 31;
  const int half = lane >> 4;
  const int l16  = lane & 15;
  const int bh   = blockIdx.x;
  const int b    = bh / H_;
  const int h    = bh % H_;
  const int i0   = blockIdx.y * 128;
  const int pm1  = pm1p[b];
  const int pm2  = pm2p[b];
  const size_t rowbase = (size_t)b * S_;
  const int colq  = h * DKV;
  const int rbase = w * 16;
  const float scale = 0.125f;                  // 1/sqrt(64)
  const size_t ktbase = (size_t)bh * DKV * S_;
  constexpr int NJT = S_ / 64;

  // stage Q tile (resident); visible after first in-loop barrier
  #pragma unroll
  for (int p = 0; p < 4; ++p) {
    int r  = (t >> 3) + 32 * p;
    int c8 = (t & 7) * 8;
    uint4 vr = *(const uint4*)&qh[(rowbase + i0 + r) * D_ + colq + c8];
    *(uint4*)&Qs[r * LDQ + c8] = vr;
  }

  float mrun[8], lrun[8];
  v8f oacc[4];
  #pragma unroll
  for (int v = 0; v < 8; ++v) { mrun[v] = -INFINITY; lrun[v] = 0.f; }
  #pragma unroll
  for (int fn = 0; fn < 4; ++fn) oacc[fn] = v8f_zero();

  if (t < 32) {
    tdm_load_2d(lds_offset(&Kts[0][0]), &khT[ktbase],                    64, 64, S_, 4, 3);
    tdm_load_2d(lds_offset(&Vs[0][0]),  &vh[rowbase * D_ + colq],        64, 64, D_, 4, 3);
  }

  for (int jt = 0; jt < NJT; ++jt) {
    const int j0  = jt * 64;
    const int buf = jt & 1;
    if (t < 32) {
      if (jt + 1 < NJT) {
        int jn = (jt + 1) * 64;
        tdm_load_2d(lds_offset(&Kts[buf ^ 1][0]), &khT[ktbase + jn],               64, 64, S_, 4, 3);
        tdm_load_2d(lds_offset(&Vs[buf ^ 1][0]),  &vh[(rowbase + jn) * D_ + colq], 64, 64, D_, 4, 3);
        __builtin_amdgcn_s_wait_tensorcnt(2);
      } else {
        __builtin_amdgcn_s_wait_tensorcnt(0);
      }
    }
    __syncthreads();

    const unsigned short* Kt = &Kts[buf][0];
    const unsigned short* Vt = &Vs[buf][0];

    // ---- S = Q K^T ----
    v8f sf[4];
    #pragma unroll
    for (int fn = 0; fn < 4; ++fn) sf[fn] = v8f_zero();
    #pragma unroll
    for (int ks = 0; ks < DKV; ks += 32) {
      v16bf a = load_a_frag(Qs, LDQ, rbase + l16, ks, half);
      #pragma unroll
      for (int fn = 0; fn < 4; ++fn) {
        v16bf bb2 = load_b_frag(&Kt[fn * 16 + l16], LDT, ks, half);
        sf[fn] = __builtin_amdgcn_wmma_f32_16x16x32_bf16(
            false, a, false, bb2, (short)0, sf[fn], false, false);
      }
    }
    // ---- scale + masks (causal, key-pad, row-pad) ----
    #pragma unroll
    for (int fn = 0; fn < 4; ++fn) {
      int j = j0 + fn * 16 + l16;
      #pragma unroll
      for (int v = 0; v < 8; ++v) {
        int i = i0 + rbase + v + 8 * half;
        float s = sf[fn][v] * scale;
        bool dead = (j > i) | (j >= pm1) | (i >= pm2);
        sf[fn][v] = dead ? NEGV : s;
      }
    }
    // ---- online softmax (reduce across the 16 lanes of the half) ----
    #pragma unroll
    for (int v = 0; v < 8; ++v) {
      float mx = fmaxf(fmaxf(sf[0][v], sf[1][v]), fmaxf(sf[2][v], sf[3][v]));
      #pragma unroll
      for (int off = 1; off < 16; off <<= 1) mx = fmaxf(mx, __shfl_xor(mx, off, 16));
      float mnew = fmaxf(mrun[v], mx);
      float corr = __expf(mrun[v] - mnew);
      float rs = 0.f;
      #pragma unroll
      for (int fn = 0; fn < 4; ++fn) {
        float p = __expf(sf[fn][v] - mnew);
        sf[fn][v] = p;
        rs += p;
      }
      #pragma unroll
      for (int off = 1; off < 16; off <<= 1) rs += __shfl_xor(rs, off, 16);
      mrun[v] = mnew;
      lrun[v] = lrun[v] * corr + rs;
      #pragma unroll
      for (int fn = 0; fn < 4; ++fn) oacc[fn][v] *= corr;
    }
    // ---- P: C-layout -> A-layout via per-wave LDS bounce ----
    unsigned short* Pw = &Ps[w * 16 * LDP];
    #pragma unroll
    for (int fn = 0; fn < 4; ++fn)
      #pragma unroll
      for (int v = 0; v < 8; ++v)
        Pw[(v + 8 * half) * LDP + fn * 16 + l16] = f2bf(sf[fn][v]);
    // ---- O += P V ----
    #pragma unroll
    for (int ks = 0; ks < 64; ks += 32) {
      v16bf a = load_a_frag(Pw, LDP, l16, ks, half);
      #pragma unroll
      for (int fn = 0; fn < 4; ++fn) {
        v16bf bb2 = load_b_frag(&Vt[fn * 16 + l16], LDV, ks, half);
        oacc[fn] = __builtin_amdgcn_wmma_f32_16x16x32_bf16(
            false, a, false, bb2, (short)0, oacc[fn], false, false);
      }
    }
    __syncthreads();                            // done with buf before overwrite
  }

  // ---- normalize + write concat-head output (bf16) ----
  #pragma unroll
  for (int fn = 0; fn < 4; ++fn) {
    int col = h * DKV + fn * 16 + l16;
    #pragma unroll
    for (int v = 0; v < 8; ++v) {
      int i = i0 + rbase + v + 8 * half;
      float val = oacc[fn][v] / lrun[v];
      ob[(rowbase + i) * D_ + col] = f2bf(val);
    }
  }
}

// ---------------------------------------------------------------------------
extern "C" void kernel_launch(void* const* d_in, const int* in_sizes, int n_in,
                              void* d_out, int out_size, void* d_ws, size_t ws_size,
                              hipStream_t stream)
{
  (void)in_sizes; (void)n_in; (void)out_size; (void)ws_size;
  const float* q   = (const float*)d_in[0];
  const float* k   = (const float*)d_in[1];
  const float* v   = (const float*)d_in[2];
  const int*   pm1 = (const int*)d_in[3];
  const int*   pm2 = (const int*)d_in[4];
  const float* Wq  = (const float*)d_in[5];
  const float* Wk  = (const float*)d_in[6];
  const float* Wv  = (const float*)d_in[7];
  const float* Wo  = (const float*)d_in[8];
  float* out = (float*)d_out;

  const size_t tokens = (size_t)B_ * S_;            // 8192
  const size_t actN   = tokens * D_;                // 8M elems
  unsigned short* ws = (unsigned short*)d_ws;
  unsigned short* qb  = ws;              ws += actN;   // bf16 activations
  unsigned short* kb  = ws;              ws += actN;
  unsigned short* vb  = ws;              ws += actN;
  unsigned short* WqT = ws;              ws += (size_t)D_ * D_;
  unsigned short* WkT = ws;              ws += (size_t)D_ * D_;
  unsigned short* WvT = ws;              ws += (size_t)D_ * D_;
  unsigned short* WoT = ws;              ws += (size_t)D_ * D_;
  unsigned short* qhB = ws;              ws += actN;
  unsigned short* khT = ws;              ws += actN;   // [B*H][DK][S]
  unsigned short* vhB = ws;              ws += actN;
  unsigned short* obB = ws;              ws += actN;

  dim3 blk(256);
  // prep: convert activations, transpose+convert weights
  int n4 = (int)(actN / 4);
  hipLaunchKernelGGL(f32_to_bf16_kernel, dim3(n4 / 256), blk, 0, stream, q, qb, n4);
  hipLaunchKernelGGL(f32_to_bf16_kernel, dim3(n4 / 256), blk, 0, stream, k, kb, n4);
  hipLaunchKernelGGL(f32_to_bf16_kernel, dim3(n4 / 256), blk, 0, stream, v, vb, n4);
  hipLaunchKernelGGL(wt_transpose_kernel, dim3(16, 16, 4), blk, 0, stream,
                     Wq, Wk, Wv, Wo, WqT, WkT, WvT, WoT);

  // projections (TDM + WMMA)
  dim3 gg(D_ / 128, (unsigned)(tokens / 128));      // (8, 64)
  hipLaunchKernelGGL((gemm_tdm_wmma<0>), gg, blk, 0, stream, qb, WqT, (void*)qhB, (int)tokens, D_, D_);
  hipLaunchKernelGGL((gemm_tdm_wmma<1>), gg, blk, 0, stream, kb, WkT, (void*)khT, (int)tokens, D_, D_);
  hipLaunchKernelGGL((gemm_tdm_wmma<0>), gg, blk, 0, stream, vb, WvT, (void*)vhB, (int)tokens, D_, D_);

  // attention
  dim3 ga(B_ * H_, S_ / 128);                       // (128, 8)
  hipLaunchKernelGGL(flash_attn_wmma, ga, blk, 0, stream, qhB, khT, vhB, pm1, pm2, obB);

  // output projection (f32 out)
  hipLaunchKernelGGL((gemm_tdm_wmma<2>), gg, blk, 0, stream, obB, WoT, (void*)out, (int)tokens, D_, D_);
}
